// RelPartialLearnableMultiHeadAttn_257698037939
// MI455X (gfx1250) — compile-verified, matrix operand order checked by reading, not run on hardware
//
#include <hip/hip_runtime.h>
#include <stdint.h>

// ---------------------------------------------------------------------------
// Transformer-XL RelPartialLearnableMultiHeadAttn forward for MI455X (gfx1250)
// - All GEMMs via v_wmma_f32_16x16x32_bf16 (wave32)
// - Flash-style fused attention (no score materialization)
// - Rel-shift folded into a sliding-window GEMM + LDS diagonal gather
// - GLOBAL_LOAD_ASYNC_TO_LDS_B128 staging (ASYNCcnt) where layout permits,
//   guarded by __has_builtin with synchronous fallback
// - Double-buffered LDS in the projection GEMMs
// ---------------------------------------------------------------------------

typedef unsigned short u16;
typedef __attribute__((ext_vector_type(16))) __bf16 v16bf;
typedef __attribute__((ext_vector_type(8)))  float  v8f;
typedef __attribute__((ext_vector_type(4)))  int    v4i;

#define QLEN      1024
#define BSZ       4
#define D_MODEL   1024
#define N_HEADS   16
#define THREE_HD  3072
#define SCALE_F   0.125f   // 1/sqrt(64)
#define NEG_BIG   (-3.0e38f)

#if __has_builtin(__builtin_amdgcn_global_load_async_to_lds_b128) && \
    __has_builtin(__builtin_amdgcn_s_wait_asynccnt)
#define HAVE_ASYNC_LDS 1
// Builtin signature (from hipcc diagnostic): (AS1 v4i*, AS3 v4i*, imm, imm)
typedef __attribute__((address_space(1))) v4i AS1v4i;
typedef __attribute__((address_space(3))) v4i AS3v4i;
#else
#define HAVE_ASYNC_LDS 0
#endif

union Frag16 {
  v16bf v;
  uint4 q[2];
  u16   h[16];
};

__device__ __forceinline__ float bf2f(u16 h) {
  unsigned u = ((unsigned)h) << 16;
  return __builtin_bit_cast(float, u);
}
// Native bf16 convert (v_cvt_*bf16_f32) instead of manual RNE bit sequence.
__device__ __forceinline__ u16 f2bf(float f) {
  __bf16 b = (__bf16)f;
  return __builtin_bit_cast(u16, b);
}
__device__ __forceinline__ v8f vzero8() {
  v8f z = {0.f, 0.f, 0.f, 0.f, 0.f, 0.f, 0.f, 0.f};
  return z;
}

// 16B global -> LDS copy; async (no VGPR round-trip, ASYNCcnt) when available.
__device__ __forceinline__ void cp_b128(void* lds, const void* g) {
#if HAVE_ASYNC_LDS
  __builtin_amdgcn_global_load_async_to_lds_b128(
      (AS1v4i*)(g), (AS3v4i*)(lds), 0, 0);
#else
  *(uint4*)lds = *(const uint4*)g;
#endif
}
__device__ __forceinline__ void wait_async0() {
#if HAVE_ASYNC_LDS
  __builtin_amdgcn_s_wait_asynccnt(0);
#endif
}

// D = A(16x32) * B(32x16) + C, bf16 in / f32 accumulate
__device__ __forceinline__ v8f wmma_bf16(const Frag16& a, const Frag16& b, v8f c) {
  return __builtin_amdgcn_wmma_f32_16x16x32_bf16(
      /*neg_a=*/false, a.v, /*neg_b=*/false, b.v,
      /*c_mod=*/(short)0, c, /*reuse_a=*/false, /*reuse_b=*/false);
}

// Load a 16x32 fragment (A layout, or B layout when `base` is [col][k]).
// lanes 0-15: rows 0..15, K halves {0..7,16..23}; lanes 16-31: K {8..15,24..31}.
__device__ __forceinline__ void load_frag(Frag16& f, const u16* base,
                                          int row16, int stride, int kbase) {
  int lane = threadIdx.x & 31;
  int hf = lane >> 4, ln = lane & 15;
  const u16* p = base + (size_t)(row16 + ln) * stride + kbase + hf * 8;
  f.q[0] = *(const uint4*)(p);
  f.q[1] = *(const uint4*)(p + 16);
}

// ---------------------------------------------------------------------------
// fp32 -> bf16 conversion
// ---------------------------------------------------------------------------
__global__ void f2bf_kernel(const float* __restrict__ in, u16* __restrict__ out, int n) {
  int i = blockIdx.x * blockDim.x + threadIdx.x;
  int stride = gridDim.x * blockDim.x;
  for (; i < n; i += stride) out[i] = f2bf(in[i]);
}

// ---------------------------------------------------------------------------
// Tiled bf16 GEMM: C[M,N] = A[M,K] * B[K,N] + bias (+ residual for fp32 out).
// Block 256 = 8 waves; 128x128 tile; wave grid 2x4, each wave 64x32 (8 C-tiles).
// Double-buffered LDS; A tile staged with async-to-LDS, B transposed on store.
// ---------------------------------------------------------------------------
template <bool FP32OUT>
__global__ __launch_bounds__(256) void gemm_bf16_kernel(
    const u16* __restrict__ A, const u16* __restrict__ B,
    const float* __restrict__ bias, const float* __restrict__ resid,
    void* __restrict__ outPtr, int M, int N, int K)
{
  __shared__ alignas(16) u16 As[2][128 * 32];   // [m][k]
  __shared__ alignas(16) u16 Bs[2][128 * 32];   // [n][k]  (transposed on store)

  const int tid = threadIdx.x;
  const int n0 = blockIdx.x * 128, m0 = blockIdx.y * 128;
  const int wv = tid >> 5, lane = tid & 31;
  const int hf = lane >> 4, ln = lane & 15;
  const int wrow = wv >> 2, wcol = wv & 3;

  auto stageA = [&](int k0, int buf) {
#pragma unroll
    for (int it = 0; it < 2; ++it) {
      int idx = tid + 256 * it;              // 0..511
      int r = idx >> 2, cq = idx & 3;
      cp_b128(&As[buf][r * 32 + cq * 8],
              &A[(size_t)(m0 + r) * K + k0 + cq * 8]);
    }
  };
  auto stageB = [&](int k0, int buf) {
#pragma unroll
    for (int it = 0; it < 2; ++it) {
      int idx = tid + 256 * it;
      int kk = idx >> 4, nq = idx & 15;
      const u16* src = &B[(size_t)(k0 + kk) * N + n0 + nq * 8];
      uint4 t = *(const uint4*)src;
      if (k0 + 32 < K) __builtin_prefetch(src + (size_t)32 * N, 0, 1);
      u16 tmp[8];
      __builtin_memcpy(tmp, &t, 16);
#pragma unroll
      for (int e = 0; e < 8; ++e) Bs[buf][(nq * 8 + e) * 32 + kk] = tmp[e];
    }
  };

  v8f acc[4][2];
#pragma unroll
  for (int rs = 0; rs < 4; ++rs)
#pragma unroll
    for (int cs = 0; cs < 2; ++cs) acc[rs][cs] = vzero8();

  // Prologue: fill buffer 0
  stageA(0, 0);
  stageB(0, 0);
  wait_async0();
  __syncthreads();

  int cur = 0;
  for (int k0 = 0; k0 < K; k0 += 32) {
    const int nxt = cur ^ 1;
    if (k0 + 32 < K) {           // stream next tile while computing this one
      stageA(k0 + 32, nxt);
      stageB(k0 + 32, nxt);
    }

    Frag16 af[4], bf[2];
#pragma unroll
    for (int rs = 0; rs < 4; ++rs) load_frag(af[rs], As[cur], wrow * 64 + rs * 16, 32, 0);
#pragma unroll
    for (int cs = 0; cs < 2; ++cs) load_frag(bf[cs], Bs[cur], wcol * 32 + cs * 16, 32, 0);
#pragma unroll
    for (int rs = 0; rs < 4; ++rs)
#pragma unroll
      for (int cs = 0; cs < 2; ++cs)
        acc[rs][cs] = wmma_bf16(af[rs], bf[cs], acc[rs][cs]);

    wait_async0();
    __syncthreads();
    cur = nxt;
  }

  // Epilogue: C layout: VGPR v -> row v + 8*hf, col = ln
#pragma unroll
  for (int rs = 0; rs < 4; ++rs)
#pragma unroll
    for (int cs = 0; cs < 2; ++cs)
#pragma unroll
      for (int v = 0; v < 8; ++v) {
        int gr = m0 + wrow * 64 + rs * 16 + v + 8 * hf;
        int gc = n0 + wcol * 32 + cs * 16 + ln;
        float val = acc[rs][cs][v] + bias[gc];
        if constexpr (FP32OUT) {
          ((float*)outPtr)[(size_t)gr * N + gc] = val + resid[(size_t)gr * N + gc];
        } else {
          ((u16*)outPtr)[(size_t)gr * N + gc] = f2bf(val);
        }
      }
}

// ---------------------------------------------------------------------------
// Fused rel-attention: per workgroup = (b, head, 64-query block), 4 waves.
// Causal j-loop; rel-shift folded: BD[i,j] = (q_i+r_r_bias) . rk[1023-(i-j)],
// computed per-wave as a 16x96 sliding-window GEMM then diagonally gathered.
// Online softmax; P@V via WMMA with P round-tripped through LDS.
// K tile and rk window staged with async-to-LDS (pure row-major copies).
// ---------------------------------------------------------------------------
__global__ __launch_bounds__(128) void attn_kernel(
    const u16* __restrict__ heads,   // [i, b, 3*H*D] bf16
    const u16* __restrict__ rk,      // [j, H*D] bf16
    const float* __restrict__ rwb,   // [H*D]
    const float* __restrict__ rrb,   // [H*D]
    u16* __restrict__ vec)           // [i, b, H*D] bf16
{
  const int i0 = blockIdx.x * 64;
  const int bn = blockIdx.y;
  const int b = bn >> 4, n = bn & 15;

  const int tid = threadIdx.x;
  const int wv = tid >> 5, lane = tid & 31;
  const int hf = lane >> 4, ln = lane & 15;
  const int iw = i0 + 16 * wv;

  __shared__ alignas(16) u16 Kt[64 * 64];      // [j][d]
  __shared__ alignas(16) u16 Vt[64 * 64];      // [d][j] (transposed)
  __shared__ alignas(16) u16 RKs[144 * 64];    // [c][d] rk sliding window
  __shared__ alignas(16) u16 BDs[4][16 * 96];  // per-wave bd scratch
  __shared__ alignas(16) u16 Ps[4][16 * 64];   // per-wave P scratch

  // Load q fragments and add biases (qw = q + r_w_bias, qr = q + r_r_bias)
  Frag16 qw[2], qr[2];
  {
    const u16* qbase = heads + (size_t)(iw * 4 + b) * THREE_HD + n * 64;
#pragma unroll
    for (int kd = 0; kd < 2; ++kd) {
      Frag16 raw;
      const u16* p = qbase + (size_t)ln * (4 * THREE_HD) + kd * 32 + hf * 8;
      raw.q[0] = *(const uint4*)p;
      raw.q[1] = *(const uint4*)(p + 16);
#pragma unroll
      for (int e = 0; e < 16; ++e) {
        int d = kd * 32 + hf * 8 + ((e < 8) ? e : (e + 8));
        float qv = bf2f(raw.h[e]);
        qw[kd].h[e] = f2bf(qv + rwb[n * 64 + d]);
        qr[kd].h[e] = f2bf(qv + rrb[n * 64 + d]);
      }
    }
  }

  v8f o[4];
#pragma unroll
  for (int ds = 0; ds < 4; ++ds) o[ds] = vzero8();
  float mrow[8], lrow[8];
#pragma unroll
  for (int v = 0; v < 8; ++v) { mrow[v] = NEG_BIG; lrow[v] = 0.0f; }

  for (int j0 = 0; j0 <= i0; j0 += 64) {
    __syncthreads();
    // ---- Stage K tile [j][d]: async row-major copy ----
#pragma unroll
    for (int it = 0; it < 4; ++it) {
      int idx = tid + 128 * it;            // 512 x b128
      int jj = idx >> 3, qd = idx & 7;
      cp_b128(&Kt[jj * 64 + qd * 8],
              &heads[(size_t)((j0 + jj) * 4 + b) * THREE_HD + 1024 + n * 64 + qd * 8]);
    }
    // ---- Stage rk window (rows ubase..ubase+143): async, source-clamped.
    //      Rows past 1023 only ever feed causally-masked scores. ----
    const int ubase = 960 - i0 + j0;
#pragma unroll
    for (int it = 0; it < 9; ++it) {
      int idx = tid + 128 * it;            // 1152 x b128
      int rr = idx >> 3, qd = idx & 7;
      int gidx = ubase + rr;
      if (gidx > 1023) gidx = 1023;
      cp_b128(&RKs[rr * 64 + qd * 8],
              &rk[(size_t)gidx * 1024 + n * 64 + qd * 8]);
    }
    // ---- Stage V tile transposed [d][j] (needs 2-byte shuffle: sync) ----
#pragma unroll
    for (int it = 0; it < 4; ++it) {
      int idx = tid + 128 * it;
      int jj = idx >> 3, qd = idx & 7;
      uint4 t = *(const uint4*)&heads[(size_t)((j0 + jj) * 4 + b) * THREE_HD + 2048 + n * 64 + qd * 8];
      u16 tmp[8];
      __builtin_memcpy(tmp, &t, 16);
#pragma unroll
      for (int e = 0; e < 8; ++e) Vt[(qd * 8 + e) * 64 + jj] = tmp[e];
    }
    wait_async0();
    __syncthreads();

    // ---- BD window GEMM: bd[16 x 96] = qr . RKs(window) ----
    const int wbase = 48 - 16 * wv;        // this wave's window offset in RKs
    {
      v8f bd[6];
#pragma unroll
      for (int cs = 0; cs < 6; ++cs) bd[cs] = vzero8();
#pragma unroll
      for (int cs = 0; cs < 6; ++cs)
#pragma unroll
        for (int kd = 0; kd < 2; ++kd) {
          Frag16 bfr;
          load_frag(bfr, RKs, wbase + cs * 16, 64, kd * 32);
          bd[cs] = wmma_bf16(qr[kd], bfr, bd[cs]);
        }
      // spill to per-wave LDS for the diagonal gather
#pragma unroll
      for (int cs = 0; cs < 6; ++cs)
#pragma unroll
        for (int v = 0; v < 8; ++v)
          BDs[wv][(v + 8 * hf) * 96 + cs * 16 + ln] = f2bf(bd[cs][v]);
    }

    // ---- AC GEMM + compose S = SCALE*(AC + BD_shifted), causal mask ----
    v8f s[4];
#pragma unroll
    for (int js = 0; js < 4; ++js) {
      s[js] = vzero8();
#pragma unroll
      for (int kd = 0; kd < 2; ++kd) {
        Frag16 kf;
        load_frag(kf, Kt, js * 16, 64, kd * 32);
        s[js] = wmma_bf16(qw[kd], kf, s[js]);
      }
#pragma unroll
      for (int v = 0; v < 8; ++v) {
        int r = v + 8 * hf;
        int gi = iw + r;
        int gj = j0 + js * 16 + ln;
        float val = s[js][v] + bf2f(BDs[wv][r * 96 + (15 - r + js * 16 + ln)]);
        val *= SCALE_F;
        if (gj > gi) val = NEG_BIG;        // causal mask
        s[js][v] = val;
      }
    }

    // ---- Online softmax update ----
#pragma unroll
    for (int v = 0; v < 8; ++v) {
      float mx = fmaxf(fmaxf(s[0][v], s[1][v]), fmaxf(s[2][v], s[3][v]));
#pragma unroll
      for (int off = 8; off >= 1; off >>= 1)
        mx = fmaxf(mx, __shfl_xor(mx, off, 16));
      float mnew = fmaxf(mrow[v], mx);
      float corr = __expf(mrow[v] - mnew);
      mrow[v] = mnew;
      float rs = 0.0f;
#pragma unroll
      for (int js = 0; js < 4; ++js) {
        float p = __expf(s[js][v] - mnew);
        s[js][v] = p;
        rs += p;
      }
#pragma unroll
      for (int off = 8; off >= 1; off >>= 1)
        rs += __shfl_xor(rs, off, 16);
      lrow[v] = lrow[v] * corr + rs;
#pragma unroll
      for (int ds = 0; ds < 4; ++ds) o[ds][v] *= corr;
    }

    // ---- P -> LDS (A layout source), then O += P @ V ----
#pragma unroll
    for (int js = 0; js < 4; ++js)
#pragma unroll
      for (int v = 0; v < 8; ++v)
        Ps[wv][(v + 8 * hf) * 64 + js * 16 + ln] = f2bf(s[js][v]);

    Frag16 pf[2];
#pragma unroll
    for (int kj = 0; kj < 2; ++kj) load_frag(pf[kj], Ps[wv], 0, 64, kj * 32);
#pragma unroll
    for (int ds = 0; ds < 4; ++ds)
#pragma unroll
      for (int kj = 0; kj < 2; ++kj) {
        Frag16 vf;
        load_frag(vf, Vt, ds * 16, 64, kj * 32);
        o[ds] = wmma_bf16(pf[kj], vf, o[ds]);
      }
  }

  // ---- Normalize and store attention vector ----
#pragma unroll
  for (int v = 0; v < 8; ++v) {
    float inv = 1.0f / lrow[v];
    int gi = iw + v + 8 * hf;
#pragma unroll
    for (int ds = 0; ds < 4; ++ds)
      vec[(size_t)(gi * 4 + b) * 1024 + n * 64 + ds * 16 + ln] = f2bf(o[ds][v] * inv);
  }
}

// ---------------------------------------------------------------------------
// LayerNorm over last dim (1024) of [4096, 1024]
// ---------------------------------------------------------------------------
__global__ __launch_bounds__(256) void ln_kernel(
    const float* __restrict__ x, const float* __restrict__ g,
    const float* __restrict__ bb, float* __restrict__ out)
{
  const int row = blockIdx.x, tid = threadIdx.x;
  __shared__ float rs[256], rq[256];
  const float* xr = x + (size_t)row * D_MODEL;
  float s = 0.f, sq = 0.f;
  for (int c = tid; c < D_MODEL; c += 256) {
    float v = xr[c];
    s += v; sq += v * v;
  }
  rs[tid] = s; rq[tid] = sq;
  __syncthreads();
  for (int off = 128; off > 0; off >>= 1) {
    if (tid < off) { rs[tid] += rs[tid + off]; rq[tid] += rq[tid + off]; }
    __syncthreads();
  }
  float mu = rs[0] * (1.0f / D_MODEL);
  float var = rq[0] * (1.0f / D_MODEL) - mu * mu;
  float inv = rsqrtf(var + 1e-5f);
  float* orow = out + (size_t)row * D_MODEL;
  for (int c = tid; c < D_MODEL; c += 256)
    orow[c] = g[c] * (xr[c] - mu) * inv + bb[c];
}

// ---------------------------------------------------------------------------
// Launch: convert -> QKV gemm -> rk gemm -> fused attention -> O gemm(+resid)
//         -> LayerNorm.  Workspace use: ~70 MB.
// ---------------------------------------------------------------------------
extern "C" void kernel_launch(void* const* d_in, const int* in_sizes, int n_in,
                              void* d_out, int out_size, void* d_ws, size_t ws_size,
                              hipStream_t stream)
{
  const float* w    = (const float*)d_in[0];   // [1024,4,1024]
  const float* r    = (const float*)d_in[1];   // [1024,1024]
  const float* rwb  = (const float*)d_in[2];   // [16,64]
  const float* rrb  = (const float*)d_in[3];   // [16,64]
  const float* qkvW = (const float*)d_in[4];   // [1024,3072]
  const float* qkvB = (const float*)d_in[5];   // [3072]
  const float* rkW  = (const float*)d_in[6];   // [1024,1024]
  const float* rkB  = (const float*)d_in[7];   // [1024]
  const float* oW   = (const float*)d_in[8];   // [1024,1024]
  const float* oB   = (const float*)d_in[9];   // [1024]
  const float* lnG  = (const float*)d_in[10];  // [1024]
  const float* lnB  = (const float*)d_in[11];  // [1024]
  // d_in[12] = attn_mask: causal triu(1) — computed analytically in-kernel.

  char* ws = (char*)d_ws;
  size_t off = 0;
  auto alloc = [&](size_t bytes) -> void* {
    void* p = ws + off;
    off += (bytes + 255) & ~(size_t)255;
    return p;
  };
  u16* wBf      = (u16*)alloc((size_t)4096 * 1024 * 2);   // 8 MB
  u16* qkvWBf   = (u16*)alloc((size_t)1024 * 3072 * 2);   // 6 MB
  u16* rkWBf    = (u16*)alloc((size_t)1024 * 1024 * 2);   // 2 MB
  u16* rBf      = (u16*)alloc((size_t)1024 * 1024 * 2);   // 2 MB
  u16* oWBf     = (u16*)alloc((size_t)1024 * 1024 * 2);   // 2 MB
  u16* headsBf  = (u16*)alloc((size_t)4096 * 3072 * 2);   // 24 MB
  u16* rkBf     = (u16*)alloc((size_t)1024 * 1024 * 2);   // 2 MB
  u16* vecBf    = (u16*)alloc((size_t)4096 * 1024 * 2);   // 8 MB
  float* attnPre = (float*)alloc((size_t)4096 * 1024 * 4); // 16 MB

  // fp32 -> bf16 conversions
  f2bf_kernel<<<2048, 256, 0, stream>>>(w, wBf, 4096 * 1024);
  f2bf_kernel<<<2048, 256, 0, stream>>>(qkvW, qkvWBf, 1024 * 3072);
  f2bf_kernel<<<1024, 256, 0, stream>>>(rkW, rkWBf, 1024 * 1024);
  f2bf_kernel<<<1024, 256, 0, stream>>>(r, rBf, 1024 * 1024);
  f2bf_kernel<<<1024, 256, 0, stream>>>(oW, oWBf, 1024 * 1024);

  // heads = w @ qkv_w + qkv_b            [4096, 3072]
  gemm_bf16_kernel<false><<<dim3(3072 / 128, 4096 / 128), 256, 0, stream>>>(
      wBf, qkvWBf, qkvB, nullptr, headsBf, 4096, 3072, 1024);
  // rk = r @ rk_w + rk_b                 [1024, 1024]
  gemm_bf16_kernel<false><<<dim3(1024 / 128, 1024 / 128), 256, 0, stream>>>(
      rBf, rkWBf, rkB, nullptr, rkBf, 1024, 1024, 1024);

  // fused relative attention -> vec      [4096, 1024]
  attn_kernel<<<dim3(QLEN / 64, BSZ * N_HEADS), 128, 0, stream>>>(
      headsBf, rkBf, rwb, rrb, vecBf);

  // attn_pre = vec @ o_w + o_b + w       [4096, 1024] fp32
  gemm_bf16_kernel<true><<<dim3(1024 / 128, 4096 / 128), 256, 0, stream>>>(
      vecBf, oWBf, oB, w, attnPre, 4096, 1024, 1024);

  // LayerNorm -> d_out
  ln_kernel<<<4096, 256, 0, stream>>>(attnPre, lnG, lnB, (float*)d_out);
}
